// AutoCorrelation_36661840839444
// MI455X (gfx1250) — matrix operand order repaired and checked by reading, not compiled, not run on hardware
//
#include <hip/hip_runtime.h>
#include <hip/hip_bf16.h>

// ---------------------------------------------------------------------------
// AutoCorrelation (Autoformer-style) for MI455X / gfx1250, wave32 + WMMA.
//   corr = (1/L) * [C | S] x [Pre ; -Pim]   (single K=1024 WMMA GEMM)
//   Pre  =  Cq*Ck + Sq*Sk,  -Pim = Sq*Ck - Cq*Sk   (DFT via WMMA GEMMs)
// GEMM tiles staged with CDNA5 async global->LDS DMA (ASYNCcnt), double-
// buffered LDS so tile k+1 DMA overlaps WMMA on tile k.  Running-pointer
// staging keeps DMA addressing to 5 pointer increments per K-step.
// ---------------------------------------------------------------------------

#define Lq 512
#define Dq 4096
#define Hq 8
#define dq 512
#define Bq 16
#define TOPK 12
#define CH_SLABS 8                 // (b,h) slabs per chunk
#define SLAB_ELEMS (Lq * dq)       // 262144

typedef _Float16 h8   __attribute__((ext_vector_type(8)));
typedef _Float16 v16h __attribute__((ext_vector_type(16)));
typedef float    v8f  __attribute__((ext_vector_type(8)));

// async copy of 16B global -> LDS, tracked by ASYNCcnt
__device__ __forceinline__ void async_b128(const void* gptr, unsigned lds_off) {
  asm volatile("global_load_async_to_lds_b128 %0, %1, off"
               :: "v"(lds_off), "v"(gptr) : "memory");
}
__device__ __forceinline__ void wait_async0() {
  asm volatile("s_wait_asynccnt 0x0" ::: "memory");
}
__device__ __forceinline__ unsigned lds_addr(const void* p) {
  return (unsigned)(uintptr_t)p;   // low 32 bits of flat addr = LDS byte addr
}

// ------------------------------- twiddles ----------------------------------
// trig[f][k]: k<512 -> cos(2*pi*f*k/512) ; k>=512 -> sin(2*pi*f*(k-512)/512)
__global__ __launch_bounds__(256) void k_trig(_Float16* __restrict__ T) {
  int e = blockIdx.x * 256 + threadIdx.x;        // 512*1024 total
  int f = e >> 10, k = e & 1023;
  int m = (f * (k & 511)) & 511;                 // exact periodic reduction
  float ang = (float)m * 1.2271846644e-2f;       // 2*pi/512
  T[e] = (_Float16)((k < 512) ? cosf(ang) : sinf(ang));
}

// ---------------- transpose+convert:  Q/K [t][c] f32 -> X [c][t] f16 -------
__global__ __launch_bounds__(256) void k_prep(const float* __restrict__ Qin,
                                              const float* __restrict__ Kin,
                                              _Float16* __restrict__ X, int slab0) {
  int z = blockIdx.z, sl = z >> 1, which = z & 1;
  int s = slab0 + sl, b = s >> 3, h = s & 7;
  const float* src = (which ? Kin : Qin) + (size_t)b * Lq * Dq + (size_t)h * dq;
  int c0 = blockIdx.x * 64, t0 = blockIdx.y * 64;
  __shared__ float tile[64 * 65];
  int tid = threadIdx.x;
  for (int i = tid; i < 4096; i += 256) {        // rows t, cols c (coalesced)
    int r = i >> 6, c = i & 63;
    tile[r * 65 + c] = src[(size_t)(t0 + r) * Dq + c0 + c];
  }
  __syncthreads();
  _Float16* dst = X + (size_t)(sl * 2 + which) * SLAB_ELEMS;   // [c][t]
  for (int i = tid; i < 2048; i += 256) {        // 64 c-rows x 32 t-pairs
    int c = i >> 5, tp = (i & 31) * 2;
    union { unsigned int u; _Float16 hh[2]; } u;
    u.hh[0] = (_Float16)tile[tp * 65 + c];
    u.hh[1] = (_Float16)tile[(tp + 1) * 65 + c];
    *(unsigned int*)&dst[(size_t)(c0 + c) * 512 + t0 + tp] = u.u;
  }
}

// ------------------------- WMMA fragment helpers ---------------------------
// LDS tiles use pitch 40 halves (80B rows: 16B-aligned, low bank conflict).
__device__ inline v16h load_a_frag(const _Float16* As, int m, int hi) {
  union { v16h v; h8 p[2]; } u;
  u.p[0] = *(const h8*)&As[m * 40 + hi * 8];
  u.p[1] = *(const h8*)&As[m * 40 + 16 + hi * 8];
  return u.v;
}
__device__ inline v16h load_b_frag(const _Float16* Bs, int n, int hi) {
  union { v16h v; h8 p[2]; } u;
  u.p[0] = *(const h8*)&Bs[n * 40 + hi * 8];
  u.p[1] = *(const h8*)&Bs[n * 40 + 16 + hi * 8];
  return u.v;
}

// ------------------------------ GEMM core ----------------------------------
// D[512m x 512n] = A[m][k] x B[n][k]^T ; A pitch 1024, B pitch BPITCH, f16.
// WG = 256 thr = 8 waves; WG tile 64m x 256n; wave tile 32m x 64n (2x4 WMMA).
// Double-buffered LDS, tiles fetched with async global->LDS DMA.
template <bool F32OUT, int KDIM, int BPITCH>
__device__ inline void gemm_core(const _Float16* __restrict__ A,
                                 const _Float16* __restrict__ B,
                                 void* __restrict__ D, float scale) {
  int m0 = blockIdx.y * 64, n0 = blockIdx.x * 256;
  __shared__ __align__(16) _Float16 As[2][64 * 40];
  __shared__ __align__(16) _Float16 Bs[2][256 * 40];

  int tid = threadIdx.x, lane = tid & 31, wid = tid >> 5;
  int laneLo = lane & 15, hi = lane >> 4;
  int wm = (wid >> 2) * 32, wn = (wid & 3) * 64;

  int arow = tid >> 2, ac = (tid & 3) * 8;       // 64 rows x 4 b128 chunks

  // running global pointers (advance by 32 halves per K-step)
  const _Float16* aptr = A + (size_t)(m0 + arow) * 1024 + ac;
  const _Float16* bp[4];
#pragma unroll
  for (int i = 0; i < 4; ++i)
    bp[i] = B + (size_t)(n0 + arow + i * 64) * BPITCH + ac;

  // fixed LDS destinations per buffer (byte offsets)
  unsigned lA[2], lB[2];
  lA[0] = lds_addr(&As[0][arow * 40 + ac]);
  lA[1] = lds_addr(&As[1][arow * 40 + ac]);
  lB[0] = lds_addr(&Bs[0][arow * 40 + ac]);      // chunk i adds i*64 rows
  lB[1] = lds_addr(&Bs[1][arow * 40 + ac]);

  v8f acc[2][4];
#pragma unroll
  for (int i = 0; i < 2; ++i)
#pragma unroll
    for (int j = 0; j < 4; ++j)
#pragma unroll
      for (int r = 0; r < 8; ++r) acc[i][j][r] = 0.f;

  auto stage = [&](int buf) {                    // issue DMA, advance pointers
    async_b128(aptr, lA[buf]);
    aptr += 32;
#pragma unroll
    for (int i = 0; i < 4; ++i) {
      async_b128(bp[i], lB[buf] + (unsigned)(i * 64 * 40 * 2));
      bp[i] += 32;
    }
  };

  stage(0);                                      // prologue
  constexpr int nsteps = KDIM >> 5;
#pragma unroll 2
  for (int st = 0; st < nsteps; ++st) {
    wait_async0();                               // my DMA for buf `st&1` done
    __syncthreads();                             // everyone's DMA visible
    if (st + 1 < nsteps) stage((st + 1) & 1);    // overlap DMA with WMMA
    int buf = st & 1;

    v16h af[2], bf[4];
#pragma unroll
    for (int mi = 0; mi < 2; ++mi)
      af[mi] = load_a_frag(As[buf], wm + mi * 16 + laneLo, hi);
#pragma unroll
    for (int ni = 0; ni < 4; ++ni)
      bf[ni] = load_b_frag(Bs[buf], wn + ni * 16 + laneLo, hi);
#pragma unroll
    for (int mi = 0; mi < 2; ++mi)
#pragma unroll
      for (int ni = 0; ni < 4; ++ni)
        acc[mi][ni] = __builtin_amdgcn_wmma_f32_16x16x32_f16(
            false, af[mi], false, bf[ni], (short)0, acc[mi][ni], false, false);
  }

#pragma unroll
  for (int mi = 0; mi < 2; ++mi)
#pragma unroll
    for (int ni = 0; ni < 4; ++ni)
#pragma unroll
      for (int r = 0; r < 8; ++r) {
        int row = m0 + wm + mi * 16 + r + 8 * hi;
        int col = n0 + wn + ni * 16 + laneLo;
        if (F32OUT)
          ((float*)D)[(size_t)row * 512 + col] = acc[mi][ni][r] * scale;
        else
          ((_Float16*)D)[(size_t)row * 512 + col] = (_Float16)acc[mi][ni][r];
      }
}

// forward DFT: z = sl*4+mat; mat: 0=Cq 1=Sq 2=Ck 3=Sk.  out f16 [f][c].
__global__ __launch_bounds__(256) void k_gemm_fwd(const _Float16* __restrict__ trig,
                                                  const _Float16* __restrict__ X,
                                                  _Float16* __restrict__ fwd) {
  int z = blockIdx.z, sl = z >> 2, mat = z & 3;
  const _Float16* A = trig + (mat & 1) * 512;                         // cos | sin
  const _Float16* B = X + (size_t)(sl * 2 + (mat >> 1)) * SLAB_ELEMS; // q | k
  _Float16* D = fwd + (size_t)(sl * 4 + mat) * SLAB_ELEMS;
  gemm_core<false, 512, 512>(A, B, D, 1.f);
}

// inverse: corr = (1/512) * [C|S](512x1024) x ppT(512c x 1024)^T, f32 [tau][c]
__global__ __launch_bounds__(256) void k_gemm_inv(const _Float16* __restrict__ trig,
                                                  const _Float16* __restrict__ ppT,
                                                  float* __restrict__ corr) {
  int sl = blockIdx.z;
  gemm_core<true, 1024, 1024>(trig, ppT + (size_t)sl * 2 * SLAB_ELEMS,
                              corr + (size_t)sl * SLAB_ELEMS, 1.f / 512.f);
}

// -------- pointwise spectrum product + transpose: ppT[c][0..511]=Pre,
//          ppT[c][512..1023] = -Pim.   grid (8 c-tiles, 8 f-tiles, CH_SLABS)
__global__ __launch_bounds__(256) void k_pwt(const _Float16* __restrict__ fwd,
                                             _Float16* __restrict__ ppT) {
  int sl = blockIdx.z, c0 = blockIdx.x * 64, f0 = blockIdx.y * 64;
  const _Float16* base = fwd + (size_t)sl * 4 * SLAB_ELEMS;
  __shared__ _Float16 tp[64 * 66];
  __shared__ _Float16 tq[64 * 66];
  int tid = threadIdx.x;
  for (int i = tid; i < 2048; i += 256) {        // 64 f-rows x 32 c-pairs
    int f = i >> 5, c2 = (i & 31) * 2;
    size_t ga = (size_t)(f0 + f) * 512 + c0 + c2;
    union { unsigned int u; _Float16 hh[2]; } a, b, c, d;
    a.u = *(const unsigned int*)&base[ga];
    b.u = *(const unsigned int*)&base[SLAB_ELEMS + ga];
    c.u = *(const unsigned int*)&base[2 * SLAB_ELEMS + ga];
    d.u = *(const unsigned int*)&base[3 * SLAB_ELEMS + ga];
#pragma unroll
    for (int j = 0; j < 2; ++j) {
      float cq = (float)a.hh[j], sq = (float)b.hh[j];
      float ck = (float)c.hh[j], sk = (float)d.hh[j];
      tp[f * 66 + c2 + j] = (_Float16)(cq * ck + sq * sk);   // Pre
      tq[f * 66 + c2 + j] = (_Float16)(sq * ck - cq * sk);   // -Pim
    }
  }
  __syncthreads();
  _Float16* o = ppT + (size_t)sl * 2 * SLAB_ELEMS;           // rows [c][1024]
  for (int i = tid; i < 2048; i += 256) {        // 64 c-rows x 32 f-pairs
    int c = i >> 5, f2 = (i & 31) * 2;
    union { unsigned int u; _Float16 hh[2]; } up, uq;
    up.hh[0] = tp[f2 * 66 + c];       up.hh[1] = tp[(f2 + 1) * 66 + c];
    uq.hh[0] = tq[f2 * 66 + c];       uq.hh[1] = tq[(f2 + 1) * 66 + c];
    size_t rbase = (size_t)(c0 + c) * 1024 + f0 + f2;
    *(unsigned int*)&o[rbase]       = up.u;
    *(unsigned int*)&o[rbase + 512] = uq.u;
  }
}

// --------------------------- top-k + softmax -------------------------------
__global__ __launch_bounds__(256) void k_topk(const float* __restrict__ corr,
                                              float* __restrict__ wsm,
                                              int* __restrict__ widx) {
  int sl = blockIdx.y, c0 = blockIdx.x * 16;
  __shared__ float lds[16 * 513];
  const float* cb = corr + (size_t)sl * SLAB_ELEMS;
  int tid = threadIdx.x, cl = tid & 15, tpo = tid >> 4;
  for (int t0 = 0; t0 < 512; t0 += 16)
    lds[cl * 513 + t0 + tpo] = cb[(size_t)(t0 + tpo) * 512 + c0 + cl];
  __syncthreads();

  int wid = tid >> 5, lane = tid & 31;
  for (int cc = 0; cc < 2; ++cc) {
    int chn = wid * 2 + cc;
    float vals[16];
    for (int j = 0; j < 16; ++j) vals[j] = lds[chn * 513 + lane + j * 32];
    float topv[TOPK]; int topi[TOPK];
    for (int it = 0; it < TOPK; ++it) {
      float bv = -3.402823466e38f; int bi = 0x7fffffff;
      for (int j = 0; j < 16; ++j)
        if (vals[j] > bv) { bv = vals[j]; bi = lane + j * 32; }
      for (int off = 16; off >= 1; off >>= 1) {
        float ov = __shfl_xor(bv, off, 32);
        int   oi = __shfl_xor(bi, off, 32);
        if (ov > bv || (ov == bv && oi < bi)) { bv = ov; bi = oi; }
      }
      topv[it] = bv; topi[it] = bi;
      if ((bi & 31) == lane) vals[bi >> 5] = -3.402823466e38f;
    }
    float mx = topv[0], sum = 0.f, ex[TOPK];
    for (int i = 0; i < TOPK; ++i) { ex[i] = expf(topv[i] - mx); sum += ex[i]; }
    float inv = 1.f / sum;
    if (lane == 0) {
      int ci = sl * 512 + c0 + chn;
      for (int i = 0; i < TOPK; ++i) {
        wsm[ci * TOPK + i]  = ex[i] * inv;
        widx[ci * TOPK + i] = topi[i];
      }
    }
  }
}

// ------------------------ gather / accumulate / output ---------------------
__global__ __launch_bounds__(256) void k_gather(const float* __restrict__ Vin,
                                                const float* __restrict__ wsm,
                                                const int* __restrict__ widx,
                                                float* __restrict__ out, int slab0) {
  int sl = blockIdx.y, s = slab0 + sl, b = s >> 3, h = s & 7;
  int c0 = blockIdx.x * 16;
  __shared__ float lds[16 * 513];
  int tid = threadIdx.x, cl = tid & 15, tpo = tid >> 4;
  const float* vb = Vin + (size_t)b * Lq * Dq + (size_t)h * dq + c0;
  for (int t0 = 0; t0 < 512; t0 += 16)
    lds[cl * 513 + t0 + tpo] = vb[(size_t)(t0 + tpo) * Dq + cl];
  __syncthreads();

  int wid = tid >> 5, lane = tid & 31;
  float acc[2][16];
  for (int cc = 0; cc < 2; ++cc) {
    int chn = wid * 2 + cc;
    int ci = sl * 512 + c0 + chn;
    for (int j = 0; j < 16; ++j) acc[cc][j] = 0.f;
    for (int i = 0; i < TOPK; ++i) {
      float wgt = wsm[ci * TOPK + i];
      int   dly = widx[ci * TOPK + i];
      for (int j = 0; j < 16; ++j) {
        int t = lane + j * 32;
        int tt = t + dly; tt = tt > 511 ? 511 : tt;   // min(idx+pos, L-1)
        acc[cc][j] += wgt * lds[chn * 513 + tt];
      }
    }
  }
  __syncthreads();
  for (int cc = 0; cc < 2; ++cc)
    for (int j = 0; j < 16; ++j)
      lds[(wid * 2 + cc) * 513 + lane + j * 32] = acc[cc][j];
  __syncthreads();
  float* ob = out + (size_t)b * Lq * Dq + (size_t)h * dq + c0;
  for (int t0 = 0; t0 < 512; t0 += 16)
    ob[(size_t)(t0 + tpo) * Dq + cl] = lds[cl * 513 + t0 + tpo];
}

// ------------------------------- launcher ----------------------------------
extern "C" void kernel_launch(void* const* d_in, const int* in_sizes, int n_in,
                              void* d_out, int out_size, void* d_ws, size_t ws_size,
                              hipStream_t stream) {
  const float* Q = (const float*)d_in[0];
  const float* K = (const float*)d_in[1];
  const float* V = (const float*)d_in[2];
  float* out = (float*)d_out;

  char* w = (char*)d_ws;
  size_t off = 0;
  _Float16* trig = (_Float16*)(w + off); off += (size_t)512 * 1024 * 2;                 // 1MB
  _Float16* X    = (_Float16*)(w + off); off += (size_t)CH_SLABS * 2 * SLAB_ELEMS * 2;  // 8.4MB
  _Float16* fwd  = (_Float16*)(w + off); off += (size_t)CH_SLABS * 4 * SLAB_ELEMS * 2;  // 16.8MB
  _Float16* ppT  = (_Float16*)(w + off); off += (size_t)CH_SLABS * 2 * SLAB_ELEMS * 2;  // 8.4MB
  float*    corr = (float*)   (w + off); off += (size_t)CH_SLABS * SLAB_ELEMS * 4;      // 8.4MB
  float*    wsm  = (float*)   (w + off); off += (size_t)CH_SLABS * 512 * TOPK * 4;
  int*      widx = (int*)     (w + off); off += (size_t)CH_SLABS * 512 * TOPK * 4;
  (void)ws_size; (void)in_sizes; (void)n_in; (void)out_size;

  k_trig<<<dim3(512 * 1024 / 256), dim3(256), 0, stream>>>(trig);

  const int nChunks = (Bq * Hq) / CH_SLABS;   // 16
  for (int ch = 0; ch < nChunks; ++ch) {
    int slab0 = ch * CH_SLABS;
    k_prep<<<dim3(8, 8, CH_SLABS * 2), dim3(256), 0, stream>>>(Q, K, X, slab0);
    k_gemm_fwd<<<dim3(2, 8, CH_SLABS * 4), dim3(256), 0, stream>>>(trig, X, fwd);
    k_pwt<<<dim3(8, 8, CH_SLABS), dim3(256), 0, stream>>>(fwd, ppT);
    k_gemm_inv<<<dim3(2, 8, CH_SLABS), dim3(256), 0, stream>>>(trig, ppT, corr);
    k_topk<<<dim3(32, CH_SLABS), dim3(256), 0, stream>>>(corr, wsm, widx);
    k_gather<<<dim3(32, CH_SLABS), dim3(256), 0, stream>>>(V, wsm, widx, out, slab0);
  }
}
